// MoleculeEncoder_47450798686658
// MI455X (gfx1250) — matrix operand order repaired and checked by reading, not compiled, not run on hardware
//
#include <hip/hip_runtime.h>

// Problem constants (match the reference file)
#define NN   100000   // nodes
#define NE   800000   // edges
#define DIN  128      // node_in_dim
#define DH   256      // hidden (= out) dim
#define NG   2048     // graphs

typedef __attribute__((ext_vector_type(16))) _Float16 v16h;
typedef __attribute__((ext_vector_type(8)))  _Float16 v8h;
typedef __attribute__((ext_vector_type(8)))  float    v8f;

// ---------------------------------------------------------------- utilities
__global__ void k_fill1(float* __restrict__ p, int n) {
  int t = blockIdx.x * blockDim.x + threadIdx.x;
  if (t < n) p[t] = 1.0f;
}

__global__ void k_zero(float* __restrict__ p, long long n) {
  long long t = (long long)blockIdx.x * blockDim.x + threadIdx.x;
  if (t < n) p[t] = 0.0f;
}

// deg[dst] += 1 per edge (self-loop contribution pre-filled by k_fill1)
__global__ void k_deg_edges(const int* __restrict__ dst, float* __restrict__ deg) {
  int t = blockIdx.x * blockDim.x + threadIdx.x;
  if (t < NE) atomicAdd(&deg[dst[t]], 1.0f);
}

// deg -> dis = rsqrt(deg); deg >= 1 always (self loops), matching
// where(deg>0, rsqrt(max(deg,1)), 0)
__global__ void k_rsqrt_inplace(float* __restrict__ p, int n) {
  int t = blockIdx.x * blockDim.x + threadIdx.x;
  if (t < n) p[t] = rsqrtf(p[t]);
}

__global__ void k_f32_to_f16(const float* __restrict__ x, _Float16* __restrict__ y,
                             long long n) {
  long long t = (long long)blockIdx.x * blockDim.x + threadIdx.x;
  if (t < n) y[t] = (_Float16)x[t];
}

// W[Kin x Kout] f32 -> Wt[Kout x Kin] f16 (transposed so the WMMA B-fragment
// load is contiguous per lane)
__global__ void k_transpose_w(const float* __restrict__ W, _Float16* __restrict__ Wt,
                              int Kin, int Kout) {
  int t = blockIdx.x * blockDim.x + threadIdx.x;
  if (t >= Kin * Kout) return;
  int k = t / Kout;
  int n = t - k * Kout;
  Wt[(size_t)n * Kin + k] = (_Float16)W[t];
}

// ---------------------------------------------------------------- WMMA GEMM
// C[nrows x Kout] (f32) = A[nrows x KIN] (f16) * Bt^T, Bt stored [Kout x KIN].
// One wave owns a full 16-row tile: it preloads ALL A fragments for the tile
// into registers (KIN/32 x v16h) and sweeps the Kout/16 column tiles reusing
// them, so A is fetched exactly once per row. K stepped by 32 via
// v_wmma_f32_16x16x32_f16 (fp32 accumulate).
template <int KIN>
__global__ void k_gemm_wmma(const _Float16* __restrict__ A,
                            const _Float16* __restrict__ Bt,
                            float* __restrict__ C,
                            int nrows, int Kout) {
  constexpr int KSTEPS = KIN / 32;
  const int lane = threadIdx.x & 31;
  const int wave = threadIdx.x >> 5;
  const int rowTile = blockIdx.x * (blockDim.x >> 5) + wave;
  if (rowTile * 16 >= nrows) return;
  const int half = lane >> 4;   // lane group 0-15 vs 16-31
  const int l15  = lane & 15;

  // A fragment rows: lane holds row (lane&15); 16 contiguous K values per
  // 32-chunk: K = half*16 + {0..15} (matches B fragment order).
  int arow = rowTile * 16 + l15;
  if (arow >= nrows) arow = nrows - 1;          // clamp; those D rows discarded
  const _Float16* aptr = A + (size_t)arow * KIN;

  v16h afrag[KSTEPS];
#pragma unroll
  for (int ks = 0; ks < KSTEPS; ++ks) {
    const int k0 = ks * 32;
    afrag[ks] = *(const v16h*)(aptr + k0 + half * 16);    // 32B aligned
  }

  const bool fullTile = (rowTile * 16 + 16) <= nrows;
  const int colTiles = Kout >> 4;
  const int crow0 = rowTile * 16 + half * 8;    // D: acc[j] -> row crow0+j

  for (int ct = 0; ct < colTiles; ++ct) {
    // B fragment: lane holds column (ct*16 + lane&15); 16 contiguous K values
    // starting at half*16 within each 32-chunk (Bt is [Kout x KIN] row-major).
    const _Float16* bptr = Bt + (size_t)(ct * 16 + l15) * KIN;
    v8f acc = {};
#pragma unroll
    for (int ks = 0; ks < KSTEPS; ++ks) {
      v16h b = *(const v16h*)(bptr + ks * 32 + half * 16);  // 32B aligned
      acc = __builtin_amdgcn_wmma_f32_16x16x32_f16(
          /*neg_a=*/false, afrag[ks], /*neg_b=*/false, b,
          /*c_mod=*/(short)0, acc, /*reuse_a=*/false, /*reuse_b=*/false);
    }
    float* cptr = C + (size_t)crow0 * Kout + ct * 16 + l15;
    if (fullTile) {                      // common path: no per-row branches
#pragma unroll
      for (int j = 0; j < 8; ++j) cptr[(size_t)j * Kout] = acc[j];
    } else {
#pragma unroll
      for (int j = 0; j < 8; ++j)
        if (crow0 + j < nrows) cptr[(size_t)j * Kout] = acc[j];
    }
  }
}

// ------------------------------------------------------- graph scatter stage
// Self-loop term doubles as the zero-init: agg[i] = hw[i] * dis[i]^2
__global__ void k_selfloop_init(const float* __restrict__ hw,
                                const float* __restrict__ dis,
                                float* __restrict__ agg, int K) {
  long long t = (long long)blockIdx.x * blockDim.x + threadIdx.x;
  const int perRow = K >> 2;
  long long total = (long long)NN * perRow;
  if (t >= total) return;
  long long i = t / perRow;
  int c4 = (int)(t - i * perRow) << 2;
  float d = dis[i];
  float w = d * d;
  float4 v = *(const float4*)(hw + i * (size_t)K + c4);
  v.x *= w; v.y *= w; v.z *= w; v.w *= w;
  *(float4*)(agg + i * (size_t)K + c4) = v;
}

// agg[dst] += hw[src] * (dis[src]*dis[dst]); one (edge, 4-feature) per thread
__global__ void k_scatter_edges(const float* __restrict__ hw,
                                const float* __restrict__ dis,
                                const int* __restrict__ src,
                                const int* __restrict__ dst,
                                float* __restrict__ agg, int K) {
  long long t = (long long)blockIdx.x * blockDim.x + threadIdx.x;
  const int perEdge = K >> 2;
  long long e = t / perEdge;
  if (e >= NE) return;
  int c4 = (int)(t - e * perEdge) << 2;
  int s = src[e], d = dst[e];
  float nrm = dis[s] * dis[d];
  float4 v = *(const float4*)(hw + (size_t)s * K + c4);
  float* o = agg + (size_t)d * K + c4;
  atomicAdd(o + 0, v.x * nrm);
  atomicAdd(o + 1, v.y * nrm);
  atomicAdd(o + 2, v.z * nrm);
  atomicAdd(o + 3, v.w * nrm);
}

// agg += bias, optional ReLU (in place), optional f16 re-encode for next GEMM
__global__ void k_bias_act(float* __restrict__ agg, const float* __restrict__ bias,
                           _Float16* __restrict__ xf16, int K, int doRelu,
                           int writeF16) {
  long long t = (long long)blockIdx.x * blockDim.x + threadIdx.x;
  if (t >= (long long)NN * K) return;
  int c = (int)(t % K);
  float v = agg[t] + bias[c];
  if (doRelu) v = fmaxf(v, 0.0f);
  agg[t] = v;
  if (writeF16) xf16[t] = (_Float16)v;
}

// ------------------------------------------------------------- mean pooling
__global__ void k_count(const int* __restrict__ batch, float* __restrict__ cnt) {
  int t = blockIdx.x * blockDim.x + threadIdx.x;
  if (t < NN) atomicAdd(&cnt[batch[t]], 1.0f);
}

__global__ void k_pool(const float* __restrict__ h, const int* __restrict__ batch,
                       float* __restrict__ out, int K) {
  long long t = (long long)blockIdx.x * blockDim.x + threadIdx.x;
  const int perRow = K >> 2;
  long long i = t / perRow;
  if (i >= NN) return;
  int c4 = (int)(t - i * perRow) << 2;
  int g = batch[i];
  float4 v = *(const float4*)(h + i * (size_t)K + c4);
  float* o = out + (size_t)g * K + c4;
  atomicAdd(o + 0, v.x);
  atomicAdd(o + 1, v.y);
  atomicAdd(o + 2, v.z);
  atomicAdd(o + 3, v.w);
}

__global__ void k_pool_div(float* __restrict__ out, const float* __restrict__ cnt,
                           int K) {
  long long t = (long long)blockIdx.x * blockDim.x + threadIdx.x;
  if (t >= (long long)NG * K) return;
  long long g = t / K;
  out[t] /= fmaxf(cnt[g], 1.0f);
}

// ------------------------------------------------------------------- launch
static inline int gblocks(long long n) { return (int)((n + 255) / 256); }

extern "C" void kernel_launch(void* const* d_in, const int* in_sizes, int n_in,
                              void* d_out, int out_size, void* d_ws, size_t ws_size,
                              hipStream_t stream) {
  (void)in_sizes; (void)n_in; (void)out_size; (void)ws_size;
  const float* x   = (const float*)d_in[0];
  const float* W1  = (const float*)d_in[1];
  const float* b1  = (const float*)d_in[2];
  const float* W2  = (const float*)d_in[3];
  const float* b2  = (const float*)d_in[4];
  const float* W3  = (const float*)d_in[5];
  const float* b3  = (const float*)d_in[6];
  const int* ei    = (const int*)d_in[7];     // [2, E] row-major
  const int* batch = (const int*)d_in[8];
  const int* src = ei;
  const int* dst = ei + NE;
  float* out = (float*)d_out;

  // carve workspace (256B-aligned regions)
  char* w = (char*)d_ws;
  size_t off = 0;
  auto carve = [&](size_t bytes) -> char* {
    char* p = w + off;
    off = (off + bytes + 255) & ~(size_t)255;
    return p;
  };
  float*    dis = (float*)carve((size_t)NN * 4);             // deg -> dis
  _Float16* Xh  = (_Float16*)carve((size_t)NN * DH * 2);     // f16 activations
  _Float16* Wt  = (_Float16*)carve((size_t)DH * DH * 2);     // transposed weights
  float*    hw  = (float*)carve((size_t)NN * DH * 4);        // GEMM output
  float*    agg = (float*)carve((size_t)NN * DH * 4);        // scatter accumulator
  float*    cnt = (float*)carve((size_t)NG * 4);

  const int B = 256;
  const int rowTiles   = (NN + 15) / 16;
  const int gemmBlocks = (rowTiles + 7) / 8;   // 8 waves (row tiles) per block

  // --- gcn_norm ---
  k_fill1<<<gblocks(NN), B, 0, stream>>>(dis, NN);
  k_deg_edges<<<gblocks(NE), B, 0, stream>>>(dst, dis);
  k_rsqrt_inplace<<<gblocks(NN), B, 0, stream>>>(dis, NN);

  // --- layer 1: x[N,128] @ W1 -> 256, scatter, +b1, ReLU ---
  k_f32_to_f16<<<gblocks((long long)NN * DIN), B, 0, stream>>>(x, Xh, (long long)NN * DIN);
  k_transpose_w<<<gblocks((long long)DIN * DH), B, 0, stream>>>(W1, Wt, DIN, DH);
  k_gemm_wmma<DIN><<<gemmBlocks, B, 0, stream>>>(Xh, Wt, hw, NN, DH);
  k_selfloop_init<<<gblocks((long long)NN * (DH / 4)), B, 0, stream>>>(hw, dis, agg, DH);
  k_scatter_edges<<<gblocks((long long)NE * (DH / 4)), B, 0, stream>>>(hw, dis, src, dst, agg, DH);
  k_bias_act<<<gblocks((long long)NN * DH), B, 0, stream>>>(agg, b1, Xh, DH, 1, 1);

  // --- layer 2 ---
  k_transpose_w<<<gblocks((long long)DH * DH), B, 0, stream>>>(W2, Wt, DH, DH);
  k_gemm_wmma<DH><<<gemmBlocks, B, 0, stream>>>(Xh, Wt, hw, NN, DH);
  k_selfloop_init<<<gblocks((long long)NN * (DH / 4)), B, 0, stream>>>(hw, dis, agg, DH);
  k_scatter_edges<<<gblocks((long long)NE * (DH / 4)), B, 0, stream>>>(hw, dis, src, dst, agg, DH);
  k_bias_act<<<gblocks((long long)NN * DH), B, 0, stream>>>(agg, b2, Xh, DH, 1, 1);

  // --- layer 3 (no ReLU) ---
  k_transpose_w<<<gblocks((long long)DH * DH), B, 0, stream>>>(W3, Wt, DH, DH);
  k_gemm_wmma<DH><<<gemmBlocks, B, 0, stream>>>(Xh, Wt, hw, NN, DH);
  k_selfloop_init<<<gblocks((long long)NN * (DH / 4)), B, 0, stream>>>(hw, dis, agg, DH);
  k_scatter_edges<<<gblocks((long long)NE * (DH / 4)), B, 0, stream>>>(hw, dis, src, dst, agg, DH);
  k_bias_act<<<gblocks((long long)NN * DH), B, 0, stream>>>(agg, b3, (_Float16*)nullptr, DH, 0, 0);

  // --- global mean pool ---
  k_zero<<<gblocks((long long)NG * DH), B, 0, stream>>>(out, (long long)NG * DH);
  k_zero<<<gblocks(NG), B, 0, stream>>>(cnt, NG);
  k_count<<<gblocks(NN), B, 0, stream>>>(batch, cnt);
  k_pool<<<gblocks((long long)NN * (DH / 4)), B, 0, stream>>>(agg, batch, out, DH);
  k_pool_div<<<gblocks((long long)NG * DH), B, 0, stream>>>(out, cnt, DH);
}
